// LSTMTagger_54915451846915
// MI455X (gfx1250) — compile-verified, matrix operand order checked by reading, not run on hardware
//
#include <hip/hip_runtime.h>
#include <hip/hip_bf16.h>
#include <stdint.h>

// ---------------------------------------------------------------- constants
#define NW   1024      // words in sentence
#define TC   16        // chars per word
#define EC   128       // char embedding dim
#define EW   300       // word embedding dim
#define HID  512       // BiLSTM output (2*H2)
#define H2   256       // hidden per direction
#define G    1024      // 4*H2 gate width
#define KW   812       // word-LSTM input dim (EW + HID)
#define KWP  832       // KW padded to multiple of 32
#define POS  48
#define NER  16

typedef __attribute__((ext_vector_type(8)))  _Float16 v8h;
typedef __attribute__((ext_vector_type(16))) _Float16 v16h;
typedef __attribute__((ext_vector_type(8)))  float    v8f;

__device__ __forceinline__ float sigmoidf_(float x) { return 1.f / (1.f + expf(-x)); }

// ---------------------------------------------------------------- WMMA frag load
// 16-bit A/B fragment for V_WMMA_F32_16X16X32_F16 (ISA 7.12.2):
// lanes 0-15 : row = lane,    K = k0+0..7  and k0+16..23
// lanes 16-31: row = lane-16, K = k0+8..15 and k0+24..31
// Source is row-major with leading dim `ld` (halfs); base pre-offset to row0.
__device__ __forceinline__ v16h load_frag16(const _Float16* __restrict__ base,
                                            int ld, int k0) {
  const int lane = threadIdx.x & 31;
  const int r  = lane & 15;
  const int kb = k0 + ((lane >> 4) << 3);
  const _Float16* p = base + (size_t)r * ld + kb;
  v8h lo = *(const v8h*)(p);        // K = kb .. kb+7     (16B aligned)
  v8h hi = *(const v8h*)(p + 16);   // K = kb+16 .. kb+23
  return __builtin_shufflevector(lo, hi, 0,1,2,3,4,5,6,7,8,9,10,11,12,13,14,15);
}

__device__ __forceinline__ v8f wmma_f16(v16h a, v16h b, v8f c) {
  return __builtin_amdgcn_wmma_f32_16x16x32_f16(false, a, false, b,
                                                (short)0, c, false, false);
}

// ---------------------------------------------------------------- WMMA GEMM
// C[M,N] = A[M,K] * B[N,K]^T.
// Register-blocked: each wave computes a 32x32 tile (2 A-frags x 2 B-frags ->
// 4 WMMAs per 8 b128 loads). Block = 8 waves in 2x4 grid -> 64 rows x 128 cols.
// Requires M % 64 == 0, N % 128 == 0 (or N % 128 covering via grid), K % 32 == 0.
// EPI 0: C = acc ; EPI 1: C += acc ;
// EPI 2: attention epilogue: per-row partial score
//        partials[row][col/16] = sum_colL aw[col]*tanh(acc + bias[col])
template <int EPI>
__global__ __launch_bounds__(256) void gemm16(
    const _Float16* __restrict__ A, const _Float16* __restrict__ B,
    float* __restrict__ C, int N, int K,
    const float* __restrict__ bias, const float* __restrict__ awv,
    float* __restrict__ partials, int ntile) {
  const int wave = threadIdx.x >> 5;
  const int lane = threadIdx.x & 31;
  const int row0 = blockIdx.x * 64  + (wave & 1) * 32;
  const int col0 = blockIdx.y * 128 + (wave >> 1) * 32;
  const _Float16* A0 = A + (size_t)row0 * K;
  const _Float16* A1 = A + (size_t)(row0 + 16) * K;
  const _Float16* B0 = B + (size_t)col0 * K;
  const _Float16* B1 = B + (size_t)(col0 + 16) * K;

  v8f acc00 = {}, acc01 = {}, acc10 = {}, acc11 = {};
  for (int k0 = 0; k0 < K; k0 += 32) {
    v16h a0 = load_frag16(A0, K, k0);
    v16h a1 = load_frag16(A1, K, k0);
    v16h b0 = load_frag16(B0, K, k0);
    v16h b1 = load_frag16(B1, K, k0);
    acc00 = wmma_f16(a0, b0, acc00);
    acc01 = wmma_f16(a0, b1, acc01);
    acc10 = wmma_f16(a1, b0, acc10);
    acc11 = wmma_f16(a1, b1, acc11);
  }

  const int colL = lane & 15;
  v8f accs[2][2] = {{acc00, acc01}, {acc10, acc11}};
#pragma unroll
  for (int rt = 0; rt < 2; ++rt) {
    const int rbase = row0 + rt * 16 + ((lane >> 4) << 3);  // VGPR j -> rbase+j
#pragma unroll
    for (int ct = 0; ct < 2; ++ct) {
      const int cb  = col0 + ct * 16;
      v8f acc = accs[rt][ct];
      if (EPI == 2) {
        const float aww = awv[cb + colL];
        const float bv  = bias[cb + colL];
#pragma unroll
        for (int j = 0; j < 8; ++j) {
          float w = aww * tanhf(acc[j] + bv);
#pragma unroll
          for (int off = 1; off < 16; off <<= 1) w += __shfl_xor(w, off, 32);
          if (colL == 0)
            partials[(size_t)(rbase + j) * ntile + (cb >> 4)] = w;
        }
      } else {
#pragma unroll
        for (int j = 0; j < 8; ++j) {
          size_t idx = (size_t)(rbase + j) * N + cb + colL;
          C[idx] = (EPI == 1) ? (C[idx] + acc[j]) : acc[j];
        }
      }
    }
  }
}

// ---------------------------------------------------------------- small kernels
__global__ void k_f32_to_f16(_Float16* dst, const float* src, int n) {
  int i = blockIdx.x * blockDim.x + threadIdx.x;
  if (i < n) dst[i] = (_Float16)src[i];
}
__global__ void k_zero_f16(_Float16* p, int n) {
  int i = blockIdx.x * blockDim.x + threadIdx.x;
  if (i < n) p[i] = (_Float16)0.f;
}
__global__ void k_zero_f32(float* p, int n) {
  int i = blockIdx.x * blockDim.x + threadIdx.x;
  if (i < n) p[i] = 0.f;
}
__global__ void k_pad_wih(_Float16* dst /*[G][KWP]*/, const float* src /*[G][KW]*/) {
  int n = blockIdx.x;
  for (int k = threadIdx.x; k < KWP; k += blockDim.x)
    dst[(size_t)n * KWP + k] = (k < KW) ? (_Float16)src[(size_t)n * KW + k]
                                        : (_Float16)0.f;
}
// ce16[t][n][e] = char_emb[char_seq[n][t]][e]  (time-major for per-step GEMMs)
__global__ void k_gather_ce(_Float16* ce16, const float* char_emb,
                            const int* char_seq) {
  int n = blockIdx.x, t = blockIdx.y, e = threadIdx.x;  // 128 threads
  int ch = char_seq[n * TC + t];
  ce16[((size_t)t * NW + n) * EC + e] = (_Float16)char_emb[(size_t)ch * EC + e];
}
// LSTM cell: z = Z[n][:] + b ; PyTorch gate order i,f,g,o
__global__ __launch_bounds__(256) void k_char_cell(
    const float* __restrict__ Z, const float* __restrict__ b,
    float* __restrict__ cstate, _Float16* __restrict__ h16,
    _Float16* __restrict__ enc16, int t, int dir) {
  int n = blockIdx.x;
  int j = threadIdx.x;  // 0..255
  const float* zr = Z + (size_t)n * G;
  float zi = zr[j]          + b[j];
  float zf = zr[j + H2]     + b[j + H2];
  float zg = zr[j + 2 * H2] + b[j + 2 * H2];
  float zo = zr[j + 3 * H2] + b[j + 3 * H2];
  float c = cstate[n * H2 + j];
  c = sigmoidf_(zf) * c + sigmoidf_(zi) * tanhf(zg);
  float h = sigmoidf_(zo) * tanhf(c);
  cstate[n * H2 + j] = c;
  h16[n * H2 + j] = (_Float16)h;
  enc16[((size_t)t * NW + n) * HID + dir * H2 + j] = (_Float16)h;
}
// softmax over t of summed score partials -> alpha[t][n]
__global__ void k_softmax_alpha(const float* __restrict__ partials,
                                float* __restrict__ alpha, int ntile) {
  int n = blockIdx.x * blockDim.x + threadIdx.x;  // 0..NW-1
  float s[TC];
  float m = -1e30f;
  for (int t = 0; t < TC; ++t) {
    const float* p = partials + (size_t)(t * NW + n) * ntile;
    float a = 0.f;
    for (int j = 0; j < ntile; ++j) a += p[j];
    s[t] = a; m = fmaxf(m, a);
  }
  float sum = 0.f;
  for (int t = 0; t < TC; ++t) { s[t] = expf(s[t] - m); sum += s[t]; }
  float inv = 1.f / sum;
  for (int t = 0; t < TC; ++t) alpha[t * NW + n] = s[t] * inv;
}
// context[n][h] = sum_t alpha[t][n] * enc[t][n][h]
__global__ void k_context(float* __restrict__ ctx, const float* __restrict__ alpha,
                          const _Float16* __restrict__ enc16) {
  int idx = blockIdx.x * blockDim.x + threadIdx.x;  // [0, NW*HID)
  int n = idx >> 9, h = idx & (HID - 1);
  float a = 0.f;
#pragma unroll
  for (int t = 0; t < TC; ++t)
    a += alpha[t * NW + n] * (float)enc16[((size_t)t * NW + n) * HID + h];
  ctx[idx] = a;
}
// xw[n][0:300]=word_emb[word_seq[n]], [300:812]=context[n], [812:832]=0
__global__ void k_build_xw(_Float16* xw, const float* word_emb,
                           const int* word_seq, const float* ctx) {
  int n = blockIdx.x;
  int widx = word_seq[n];
  for (int k = threadIdx.x; k < KWP; k += blockDim.x) {
    float v = 0.f;
    if (k < EW)       v = word_emb[(size_t)widx * EW + k];
    else if (k < KW)  v = ctx[(size_t)n * HID + (k - EW)];
    xw[(size_t)n * KWP + k] = (_Float16)v;
  }
}
// Word BiLSTM recurrence: one persistent block per direction; h,c,z in LDS;
// Whh streamed as f16 (512 KB/dir, L2-resident) each step -- this stream is the
// per-step latency floor of the only serial phase, so halving it matters.
__global__ __launch_bounds__(1024) void k_word_lstm(
    const float* __restrict__ Zxf, const float* __restrict__ Zxb,
    const _Float16* __restrict__ Whhf, const _Float16* __restrict__ Whhb,
    const float* __restrict__ bf, const float* __restrict__ bb,
    float* __restrict__ out) {
  const int dir = blockIdx.x;
  const float* Zx      = dir ? Zxb  : Zxf;
  const _Float16* Whh  = dir ? Whhb : Whhf;
  const float* b       = dir ? bb   : bf;
  __shared__ float h[H2], c[H2], z[G];
  const int tid = threadIdx.x;  // 0..1023
  if (tid < H2) { h[tid] = 0.f; c[tid] = 0.f; }
  __syncthreads();
  const _Float16* wrow = Whh + (size_t)tid * H2;
  const float bias = b[tid];
  for (int s = 0; s < NW; ++s) {
    const int t = dir ? (NW - 1 - s) : s;
    float acc = Zx[(size_t)t * G + tid] + bias;
#pragma unroll 8
    for (int k = 0; k < H2; ++k) acc += (float)wrow[k] * h[k];
    z[tid] = acc;
    __syncthreads();
    if (tid < H2) {
      float ig = sigmoidf_(z[tid]);
      float fg = sigmoidf_(z[tid + H2]);
      float gg = tanhf(z[tid + 2 * H2]);
      float og = sigmoidf_(z[tid + 3 * H2]);
      float cn = fg * c[tid] + ig * gg;
      float hn = og * tanhf(cn);
      c[tid] = cn; h[tid] = hn;
      out[(size_t)t * HID + dir * H2 + tid] = hn;
    }
    __syncthreads();
  }
}
// Output heads + log_softmax; one block per word, 64 threads (48 pos + 16 ner)
__global__ __launch_bounds__(64) void k_heads(
    const float* __restrict__ out512, const float* __restrict__ hpW,
    const float* __restrict__ hpb, const float* __restrict__ hnW,
    const float* __restrict__ hnb, float* __restrict__ dout) {
  __shared__ float o[HID];
  __shared__ float logits[64];
  __shared__ float stats[4];
  const int n = blockIdx.x, tid = threadIdx.x;
  for (int k = tid; k < HID; k += 64) o[k] = out512[(size_t)n * HID + k];
  __syncthreads();
  float l;
  if (tid < POS) {
    l = hpb[tid];
    const float* wr = hpW + (size_t)tid * HID;
    for (int k = 0; k < HID; ++k) l += wr[k] * o[k];
  } else {
    int q = tid - POS;
    l = hnb[q];
    const float* wr = hnW + (size_t)q * HID;
    for (int k = 0; k < HID; ++k) l += wr[k] * o[k];
  }
  logits[tid] = l;
  __syncthreads();
  if (tid == 0) {
    float m = -1e30f; for (int k = 0; k < POS; ++k) m = fmaxf(m, logits[k]);
    float s = 0.f;    for (int k = 0; k < POS; ++k) s += expf(logits[k] - m);
    stats[0] = m; stats[1] = logf(s);
  } else if (tid == 1) {
    float m = -1e30f; for (int k = POS; k < 64; ++k) m = fmaxf(m, logits[k]);
    float s = 0.f;    for (int k = POS; k < 64; ++k) s += expf(logits[k] - m);
    stats[2] = m; stats[3] = logf(s);
  }
  __syncthreads();
  if (tid < POS) dout[(size_t)n * POS + tid] = logits[tid] - stats[0] - stats[1];
  else dout[(size_t)NW * POS + (size_t)n * NER + (tid - POS)] =
           logits[tid] - stats[2] - stats[3];
}

// ---------------------------------------------------------------- launch
extern "C" void kernel_launch(void* const* d_in, const int* in_sizes, int n_in,
                              void* d_out, int out_size, void* d_ws, size_t ws_size,
                              hipStream_t stream) {
  (void)in_sizes; (void)n_in; (void)out_size; (void)ws_size;
  const float* char_emb = (const float*)d_in[0];
  const float* word_emb = (const float*)d_in[1];
  const float* cWih_f = (const float*)d_in[2];
  const float* cWhh_f = (const float*)d_in[3];
  const float* cb_f   = (const float*)d_in[4];
  const float* cWih_b = (const float*)d_in[5];
  const float* cWhh_b = (const float*)d_in[6];
  const float* cb_b   = (const float*)d_in[7];
  const float* wWih_f = (const float*)d_in[8];
  const float* wWhh_f = (const float*)d_in[9];
  const float* wb_f   = (const float*)d_in[10];
  const float* wWih_b = (const float*)d_in[11];
  const float* wWhh_b = (const float*)d_in[12];
  const float* wb_b   = (const float*)d_in[13];
  const float* aW     = (const float*)d_in[14];
  const float* ab     = (const float*)d_in[15];
  const float* aw_v   = (const float*)d_in[16];
  const float* hpW    = (const float*)d_in[17];
  const float* hpb    = (const float*)d_in[18];
  const float* hnW    = (const float*)d_in[19];
  const float* hnb    = (const float*)d_in[20];
  const int* word_seq = (const int*)d_in[21];
  const int* char_seq = (const int*)d_in[22];
  float* out = (float*)d_out;

  // ---- workspace carve (256B aligned); total ~53 MB
  uint8_t* wp = (uint8_t*)d_ws;
  auto take = [&](size_t bytes) -> void* {
    void* p = (void*)wp; wp += (bytes + 255) & ~(size_t)255; return p;
  };
  _Float16* cWih16f  = (_Float16*)take((size_t)G * EC * 2);
  _Float16* cWih16b  = (_Float16*)take((size_t)G * EC * 2);
  _Float16* cWhh16f  = (_Float16*)take((size_t)G * H2 * 2);
  _Float16* cWhh16b  = (_Float16*)take((size_t)G * H2 * 2);
  _Float16* wWhh16f  = (_Float16*)take((size_t)G * H2 * 2);
  _Float16* wWhh16b  = (_Float16*)take((size_t)G * H2 * 2);
  _Float16* aW16     = (_Float16*)take((size_t)HID * HID * 2);
  _Float16* wWihp16f = (_Float16*)take((size_t)G * KWP * 2);
  _Float16* wWihp16b = (_Float16*)take((size_t)G * KWP * 2);
  _Float16* ce16     = (_Float16*)take((size_t)TC * NW * EC * 2);
  _Float16* enc16    = (_Float16*)take((size_t)TC * NW * HID * 2);
  _Float16* h16f     = (_Float16*)take((size_t)NW * H2 * 2);
  _Float16* h16b     = (_Float16*)take((size_t)NW * H2 * 2);
  float*    cstf     = (float*)take((size_t)NW * H2 * 4);
  float*    cstb     = (float*)take((size_t)NW * H2 * 4);
  float*    Zf       = (float*)take((size_t)NW * G * 4);
  float*    Zb       = (float*)take((size_t)NW * G * 4);
  float*    partials = (float*)take((size_t)TC * NW * (HID / 16) * 4);
  float*    alpha    = (float*)take((size_t)TC * NW * 4);
  float*    ctx      = (float*)take((size_t)NW * HID * 4);
  _Float16* xw16     = (_Float16*)take((size_t)NW * KWP * 2);
  float*    Zwf      = (float*)take((size_t)NW * G * 4);
  float*    Zwb      = (float*)take((size_t)NW * G * 4);
  float*    wout     = (float*)take((size_t)NW * HID * 4);

  auto conv = [&](_Float16* dst, const float* src, int n) {
    k_f32_to_f16<<<(n + 255) / 256, 256, 0, stream>>>(dst, src, n);
  };

  // ---- weight prep (f32 -> f16, pad word Wih to K=832)
  conv(cWih16f, cWih_f, G * EC);
  conv(cWih16b, cWih_b, G * EC);
  conv(cWhh16f, cWhh_f, G * H2);
  conv(cWhh16b, cWhh_b, G * H2);
  conv(wWhh16f, wWhh_f, G * H2);
  conv(wWhh16b, wWhh_b, G * H2);
  conv(aW16, aW, HID * HID);
  k_pad_wih<<<G, 256, 0, stream>>>(wWihp16f, wWih_f);
  k_pad_wih<<<G, 256, 0, stream>>>(wWihp16b, wWih_b);

  // ---- embeddings + state init (re-zeroed every call: state is mutated)
  k_gather_ce<<<dim3(NW, TC), EC, 0, stream>>>(ce16, char_emb, char_seq);
  k_zero_f16<<<(NW * H2 + 255) / 256, 256, 0, stream>>>(h16f, NW * H2);
  k_zero_f16<<<(NW * H2 + 255) / 256, 256, 0, stream>>>(h16b, NW * H2);
  k_zero_f32<<<(NW * H2 + 255) / 256, 256, 0, stream>>>(cstf, NW * H2);
  k_zero_f32<<<(NW * H2 + 255) / 256, 256, 0, stream>>>(cstb, NW * H2);

  // ---- char BiLSTM: per step, Z = ce[t]@Wih^T (WMMA) ; Z += h@Whh^T (WMMA) ; cell
  const dim3 gZ(NW / 64, G / 128);           // 16 x 8 blocks, 64x128 tile each
  for (int s = 0; s < TC; ++s) {
    const int tf = s, tb = TC - 1 - s;
    gemm16<0><<<gZ, 256, 0, stream>>>(ce16 + (size_t)tf * NW * EC, cWih16f, Zf,
                                      G, EC, nullptr, nullptr, nullptr, 0);
    gemm16<1><<<gZ, 256, 0, stream>>>(h16f, cWhh16f, Zf,
                                      G, H2, nullptr, nullptr, nullptr, 0);
    k_char_cell<<<NW, 256, 0, stream>>>(Zf, cb_f, cstf, h16f, enc16, tf, 0);

    gemm16<0><<<gZ, 256, 0, stream>>>(ce16 + (size_t)tb * NW * EC, cWih16b, Zb,
                                      G, EC, nullptr, nullptr, nullptr, 0);
    gemm16<1><<<gZ, 256, 0, stream>>>(h16b, cWhh16b, Zb,
                                      G, H2, nullptr, nullptr, nullptr, 0);
    k_char_cell<<<NW, 256, 0, stream>>>(Zb, cb_b, cstb, h16b, enc16, tb, 1);
  }

  // ---- attention: fused energy GEMM -> tanh -> aw-dot partials (no 32MB tensor)
  gemm16<2><<<dim3(TC * NW / 64, HID / 128), 256, 0, stream>>>(
      enc16, aW16, nullptr, HID, HID, ab, aw_v, partials, HID / 16);
  k_softmax_alpha<<<NW / 256, 256, 0, stream>>>(partials, alpha, HID / 16);
  k_context<<<NW * HID / 256, 256, 0, stream>>>(ctx, alpha, enc16);

  // ---- word BiLSTM: input projection via WMMA, recurrence persistent in LDS
  k_build_xw<<<NW, 256, 0, stream>>>(xw16, word_emb, word_seq, ctx);
  gemm16<0><<<gZ, 256, 0, stream>>>(xw16, wWihp16f, Zwf,
                                    G, KWP, nullptr, nullptr, nullptr, 0);
  gemm16<0><<<gZ, 256, 0, stream>>>(xw16, wWihp16b, Zwb,
                                    G, KWP, nullptr, nullptr, nullptr, 0);
  k_word_lstm<<<2, 1024, 0, stream>>>(Zwf, Zwb, wWhh16f, wWhh16b, wb_f, wb_b, wout);

  // ---- heads + log_softmax
  k_heads<<<NW, 64, 0, stream>>>(wout, hpW, hpb, hnW, hnb, out);
}
